// MultiHeadAttentionWrapper_78254304133329
// MI455X (gfx1250) — matrix operand order repaired and checked by reading, not compiled, not run on hardware
//
#include <hip/hip_runtime.h>
#include <hip/hip_bf16.h>

typedef __attribute__((ext_vector_type(16))) __bf16 v16bf;
typedef __attribute__((ext_vector_type(8)))  float  v8f;

#define D_IN  768
#define D_OUT 64
#define NH    12
#define BB    8
#define SS    1024
#define NKC   (D_IN / 32)   // 24 K-chunks

// A-fragment (16-bit, 16x32) pair base: VGPR j holds K = 2j,2j+1 (contiguous),
// upper half-wave adds +16.
__device__ __forceinline__ int pairK(int j) { return 2 * j; }

__device__ __forceinline__ float rowmax16(float v) {
    v = fmaxf(v, __shfl_xor(v, 1, 32));
    v = fmaxf(v, __shfl_xor(v, 2, 32));
    v = fmaxf(v, __shfl_xor(v, 4, 32));
    v = fmaxf(v, __shfl_xor(v, 8, 32));
    return v;
}
__device__ __forceinline__ float rowsum16(float v) {
    v += __shfl_xor(v, 1, 32);
    v += __shfl_xor(v, 2, 32);
    v += __shfl_xor(v, 4, 32);
    v += __shfl_xor(v, 8, 32);
    return v;
}

// CDNA5 async copy: 16 bytes global -> LDS, no VGPR data, tracked by ASYNCcnt.
// LDS address = low 32 bits of the generic shared pointer.
__device__ __forceinline__ void async_copy16(unsigned lds_addr, const void* gaddr) {
    asm volatile("global_load_async_to_lds_b128 %0, %1, off"
                 :: "v"(lds_addr), "v"(gaddr) : "memory");
}
__device__ __forceinline__ unsigned lds_off(const void* p) {
    return (unsigned)(size_t)p;
}

// ---------------- Phase 0: pre-convert W to bf16, LDS-ready tiling ----------------
// wt layout: tile[(z*NH+h)*NKC + kc] = 64x32 bf16 [n][k], contiguous 4KB per tile.
// grid: (NKC, NH, 3), block 256.
__global__ __launch_bounds__(256) void wconv_kernel(
    const float* __restrict__ Wq, const float* __restrict__ Wk,
    const float* __restrict__ Wv, __bf16* __restrict__ wt)
{
    const int kc = blockIdx.x, h = blockIdx.y, z = blockIdx.z;
    const float* W = (z == 0) ? Wq : (z == 1) ? Wk : Wv;
    __bf16* dst = wt + ((size_t)(z * NH + h) * NKC + kc) * 2048;
    #pragma unroll
    for (int t = 0; t < 8; ++t) {
        int i = t * 256 + threadIdx.x;      // i = k*64 + n  (n fast -> coalesced reads)
        int k = i >> 6, n = i & 63;
        dst[n * 32 + k] = (__bf16)W[((size_t)h * D_IN + kc * 32 + k) * 64 + n];
    }
}

// ---------------- Phase 1: QKV projection ----------------
// grid: (64, 12, 3), block 256 (8 waves). Wave handles 16 rows x 64 cols, K-loop over 768.
// q,k stored [b][h][s][64]; v stored TRANSPOSED [b][h][e][s] so phase 2 needs no LDS transpose.
__global__ __launch_bounds__(256) void qkv_proj_kernel(
    const float* __restrict__ x, const __bf16* __restrict__ wt,
    __bf16* __restrict__ q_ws, __bf16* __restrict__ k_ws, __bf16* __restrict__ v_ws)
{
    __shared__ __align__(32) __bf16 lds_w[64 * 32];   // one W tile: [n][k]
    const int tid  = threadIdx.x;
    const int lane = tid & 31;
    const int w    = tid >> 5;
    const int m    = lane & 15;
    const int half = lane >> 4;
    const int h    = blockIdx.y;
    const int z    = blockIdx.z;

    __bf16* dst = (z == 0) ? q_ws : (z == 1) ? k_ws : v_ws;
    const bool vtrans = (z == 2);
    const char* wtile = (const char*)(wt + (size_t)(z * NH + h) * NKC * 2048);
    const unsigned aw_addr = lds_off(lds_w) + tid * 16;

    const int rowbase = blockIdx.x * 128 + w * 16;          // global row in [0, B*S)
    const float* xrow = x + (size_t)(rowbase + m) * D_IN;   // this lane's A row

    v8f c[4] = {};

    for (int kc = 0; kc < NKC; ++kc) {
        __syncthreads();                       // all waves done reading previous tile
        // async stage W tile: 4KB contiguous, one 16B chunk per thread
        async_copy16(aw_addr, wtile + (size_t)kc * 4096 + tid * 16);
        asm volatile("s_wait_asynccnt 0" ::: "memory");
        __syncthreads();                       // all chunks landed

        const int kb = kc * 32;
        if (kc + 1 < NKC) __builtin_prefetch(xrow + kb + 32, 0, 0);

        // A fragment (x tile, fp32 -> bf16 on the fly)
        v16bf a;
        #pragma unroll
        for (int j = 0; j < 8; ++j) {
            int kp = pairK(j) + half * 16;
            float2 xv = *(const float2*)(xrow + kb + kp);
            a[2 * j]     = (__bf16)xv.x;
            a[2 * j + 1] = (__bf16)xv.y;
        }

        // 4 N-tiles of 16: B fragment = 16 contiguous bf16 from lds_w
        #pragma unroll
        for (int f = 0; f < 4; ++f) {
            const v16bf bw = *(const v16bf*)(lds_w + (f * 16 + m) * 32 + half * 16);
            c[f] = __builtin_amdgcn_wmma_f32_16x16x32_bf16(
                false, a, false, bw, (short)0, c[f], false, false);
        }
    }

    // store C -> bf16 workspace
    if (vtrans) {
        // V: [b][h][e][s]; consecutive r -> consecutive s: pack pairs into b32 stores
        #pragma unroll
        for (int f = 0; f < 4; ++f) {
            int n = f * 16 + m;
            #pragma unroll
            for (int r = 0; r < 8; r += 2) {
                int row = rowbase + r + half * 8;
                int b = row >> 10, s = row & 1023;
                size_t idx = ((size_t)(b * NH + h) * 64 + n) * SS + s;
                unsigned u = (unsigned)__builtin_bit_cast(unsigned short, (__bf16)c[f][r])
                           | ((unsigned)__builtin_bit_cast(unsigned short, (__bf16)c[f][r + 1]) << 16);
                *(unsigned*)(dst + idx) = u;
            }
        }
    } else {
        #pragma unroll
        for (int f = 0; f < 4; ++f) {
            #pragma unroll
            for (int r = 0; r < 8; ++r) {
                int row = rowbase + r + half * 8;
                int b = row >> 10, s = row & 1023;
                dst[((size_t)(b * NH + h) * SS + s) * 64 + f * 16 + m] = (__bf16)c[f][r];
            }
        }
    }
}

// ---------------- Phase 2: causal flash attention ----------------
// grid: (16, 12, 8), block 128 (4 waves). Wave owns 16 query rows of one (b,h).
__global__ __launch_bounds__(128) void attn_kernel(
    const __bf16* __restrict__ q_ws, const __bf16* __restrict__ k_ws,
    const __bf16* __restrict__ v_ws, float* __restrict__ out)
{
    // per-wave: q 16x64 | k 32x64 [key][e] | v 64x32 [e][key] | p 16x32
    __shared__ __align__(32) __bf16 smem[4 * 5632];
    const int tid  = threadIdx.x;
    const int lane = tid & 31;
    const int w    = tid >> 5;
    const int m    = lane & 15;
    const int half = lane >> 4;
    const int h    = blockIdx.y;
    const int b    = blockIdx.z;
    const int bh   = b * NH + h;
    const int qbase = blockIdx.x * 64 + w * 16;

    __bf16* lds   = smem + w * 5632;
    __bf16* lds_q = lds;
    __bf16* lds_k = lds + 1024;
    __bf16* lds_v = lds + 3072;
    __bf16* lds_p = lds + 5120;
    const unsigned aq_addr = lds_off(lds_q);
    const unsigned ak_addr = lds_off(lds_k);
    const unsigned av_addr = lds_off(lds_v);

    // stage Q tile: 2048B contiguous, async memory->LDS
    {
        const char* g = (const char*)(q_ws + ((size_t)bh * SS + qbase) * 64) + lane * 16;
        #pragma unroll
        for (int t = 0; t < 4; ++t)
            async_copy16(aq_addr + lane * 16 + t * 512, g + t * 512);
    }
    asm volatile("s_wait_asynccnt 0" ::: "memory");

    // Q A-fragments for e-chunks 0..31 and 32..63 (kept live across key loop)
    v16bf aq0, aq1;
    #pragma unroll
    for (int j = 0; j < 8; ++j) {
        int kp = pairK(j) + half * 16;
        aq0[2 * j]     = lds_q[m * 64 + kp];
        aq0[2 * j + 1] = lds_q[m * 64 + kp + 1];
        aq1[2 * j]     = lds_q[m * 64 + 32 + kp];
        aq1[2 * j + 1] = lds_q[m * 64 + 32 + kp + 1];
    }

    float mrow[8], lrow[8];
    #pragma unroll
    for (int r = 0; r < 8; ++r) { mrow[r] = -3.0e38f; lrow[r] = 0.0f; }
    v8f acc[4] = {};
    const float SCALE = 0.125f * 1.4426950408889634f;   // 1/sqrt(64) * log2(e)

    const char* kg_base = (const char*)(k_ws + (size_t)bh * SS * 64);
    const __bf16* vg_base = v_ws + (size_t)bh * 64 * SS;   // [e][s]

    for (int kb = 0; kb < qbase + 16; kb += 32) {
        // WAR guard: previous iteration's ds_load fragment reads must drain before
        // the async engine overwrites the K/V tiles (ASYNCcnt and DScnt are unordered).
        asm volatile("s_wait_dscnt 0" ::: "memory");

        // async stage K tile [key][e]: 4096B contiguous
        {
            const char* g = kg_base + (size_t)kb * 128 + lane * 16;
            #pragma unroll
            for (int t = 0; t < 8; ++t)
                async_copy16(ak_addr + lane * 16 + t * 512, g + t * 512);
        }
        // async stage V tile [e][key]: 64 rows of 64B from the transposed V layout
        #pragma unroll
        for (int t = 0; t < 4; ++t) {
            int cidx = t * 32 + lane;       // 128 x 16B chunks
            int e = cidx >> 2, part = cidx & 3;
            async_copy16(av_addr + (e * 32 + part * 8) * 2,
                         vg_base + (size_t)e * SS + kb + part * 8);
        }
        asm volatile("s_wait_asynccnt 0" ::: "memory");

        // scores: two 16x16 subtiles, K-loop over e (2 x K=32)
        v8f sc[2] = {};
        #pragma unroll
        for (int kt = 0; kt < 2; ++kt) {
            #pragma unroll
            for (int c = 0; c < 2; ++c) {
                const v16bf bk =
                    *(const v16bf*)(lds_k + (kt * 16 + m) * 64 + c * 32 + half * 16);
                sc[kt] = __builtin_amdgcn_wmma_f32_16x16x32_bf16(
                    false, (c == 0) ? aq0 : aq1, false, bk, (short)0, sc[kt], false, false);
            }
        }

        // online softmax over both subtiles, write P tile (bf16) to LDS
        #pragma unroll
        for (int r = 0; r < 8; ++r) {
            int mg = qbase + r + half * 8;                 // global query row
            float s0 = ((kb + m)      <= mg) ? sc[0][r] * SCALE : -3.0e38f;
            float s1 = ((kb + 16 + m) <= mg) ? sc[1][r] * SCALE : -3.0e38f;
            float tm   = rowmax16(fmaxf(s0, s1));
            float mnew = fmaxf(mrow[r], tm);
            float corr = __builtin_exp2f(mrow[r] - mnew);
            float p0   = __builtin_exp2f(s0 - mnew);
            float p1   = __builtin_exp2f(s1 - mnew);
            lrow[r] = lrow[r] * corr + rowsum16(p0 + p1);
            mrow[r] = mnew;
            #pragma unroll
            for (int f = 0; f < 4; ++f) acc[f][r] *= corr;
            int mm = r + half * 8;
            lds_p[mm * 32 + m]      = (__bf16)p0;
            lds_p[mm * 32 + 16 + m] = (__bf16)p1;
        }
        asm volatile("s_wait_dscnt 0" ::: "memory");   // P RAW (cross-lane through LDS)

        // P A-fragment (16x32 over key), then acc += P x V (4 e-tiles)
        v16bf ap;
        #pragma unroll
        for (int j = 0; j < 8; ++j) {
            int kp = pairK(j) + half * 16;
            ap[2 * j]     = lds_p[m * 32 + kp];
            ap[2 * j + 1] = lds_p[m * 32 + kp + 1];
        }
        #pragma unroll
        for (int f = 0; f < 4; ++f) {
            const v16bf bv = *(const v16bf*)(lds_v + (f * 16 + m) * 32 + half * 16);
            acc[f] = __builtin_amdgcn_wmma_f32_16x16x32_bf16(
                false, ap, false, bv, (short)0, acc[f], false, false);
        }
    }

    // epilogue: normalize and store fp32 output [b][s][h*64+e]
    #pragma unroll
    for (int r = 0; r < 8; ++r) {
        float inv = 1.0f / lrow[r];
        int srow  = qbase + r + half * 8;
        float* orow = out + ((size_t)b * SS + srow) * (NH * D_OUT) + h * 64;
        #pragma unroll
        for (int f = 0; f < 4; ++f) orow[f * 16 + m] = acc[f][r] * inv;
    }
}

extern "C" void kernel_launch(void* const* d_in, const int* in_sizes, int n_in,
                              void* d_out, int out_size, void* d_ws, size_t ws_size,
                              hipStream_t stream) {
    const float* x  = (const float*)d_in[0];
    const float* Wq = (const float*)d_in[1];
    const float* Wk = (const float*)d_in[2];
    const float* Wv = (const float*)d_in[3];

    const size_t n = (size_t)BB * NH * SS * 64;   // elements per projection
    __bf16* q_ws = (__bf16*)d_ws;
    __bf16* k_ws = q_ws + n;
    __bf16* v_ws = k_ws + n;                      // stored [b][h][e][s]
    __bf16* wt_ws = v_ws + n;                     // 3*NH*NKC tiles of 2048 bf16
    float*  out  = (float*)d_out;

    dim3 g0(NKC, NH, 3), b0(256);
    wconv_kernel<<<g0, b0, 0, stream>>>(Wq, Wk, Wv, wt_ws);

    dim3 g1(64, NH, 3), b1(256);
    qkv_proj_kernel<<<g1, b1, 0, stream>>>(x, wt_ws, q_ws, k_ws, v_ws);

    dim3 g2(16, NH, BB), b2(128);
    attn_kernel<<<g2, b2, 0, stream>>>(q_ws, k_ws, v_ws, out);
}
